// Retriever_72988674228567
// MI455X (gfx1250) — matrix operand order repaired and checked by reading, not compiled, not run on hardware
//
#include <hip/hip_runtime.h>
#include <hip/hip_bf16.h>

// ---------------------------------------------------------------------------
// Retriever kernel for MI455X (gfx1250, wave32, WMMA).
//
// Factorization:  relu([q|he[h]|hr|he[t]] @ W1 + b1) @ W2 + b2
//   = relu( (q@W1q + b1) + he[h]@W1h + hr@W1r + he[t]@W1t ) @ W2 + b2
// -> per-node projections PH/PT via bf16 WMMA GEMMs, per-relation PR2
//    (with q-term + b1 folded in), then a memory-bound per-edge pass that
//    stays in the 192MB L2 (PH+PT = 102MB).
// ---------------------------------------------------------------------------

typedef __attribute__((ext_vector_type(16))) __bf16 v16bf;
typedef __attribute__((ext_vector_type(8)))  float  v8f;

#define EMB 128
#define FEAT_COLS 16     // topic(2) + fwd(6) + rev(6) + zero pad(2)
#define KREAL 142        // 128 emb + 14 feature cols
#define KPAD 160         // padded to 5 * 32
#define KSTEPS 5

// ---------------------------------------------------------------- small prep
__global__ __launch_bounds__(256)
void init_feat_kernel(const float* __restrict__ topic, float* __restrict__ Feat, int N) {
    int n = blockIdx.x * 256 + threadIdx.x;
    if (n >= N) return;
    Feat[(long)n * FEAT_COLS + 0]  = topic[2 * n + 0];
    Feat[(long)n * FEAT_COLS + 1]  = topic[2 * n + 1];
    Feat[(long)n * FEAT_COLS + 14] = 0.0f;   // K padding cols read by GEMM
    Feat[(long)n * FEAT_COLS + 15] = 0.0f;
}

__global__ __launch_bounds__(256)
void count_kernel(const int* __restrict__ h, const int* __restrict__ t,
                  float* __restrict__ cntF, float* __restrict__ cntR, int E) {
    int e = blockIdx.x * 256 + threadIdx.x;
    if (e >= E) return;
    atomicAdd(cntF + t[e], 1.0f);   // in-degree along h->t
    atomicAdd(cntR + h[e], 1.0f);   // in-degree along t->h
}

__global__ __launch_bounds__(256)
void scatter_kernel(const int* __restrict__ src, const int* __restrict__ dst,
                    const float* __restrict__ Feat, float* __restrict__ acc,
                    int incol, int E) {
    int e = blockIdx.x * 256 + threadIdx.x;
    if (e >= E) return;
    int s = src[e], d = dst[e];
    float x0 = Feat[(long)s * FEAT_COLS + incol + 0];
    float x1 = Feat[(long)s * FEAT_COLS + incol + 1];
    atomicAdd(acc + 2 * (long)d + 0, x0);
    atomicAdd(acc + 2 * (long)d + 1, x1);
}

__global__ __launch_bounds__(256)
void finalize_kernel(const float* __restrict__ acc, const float* __restrict__ cnt,
                     float* __restrict__ Feat, int outcol, int N) {
    int n = blockIdx.x * 256 + threadIdx.x;
    if (n >= N) return;
    float inv = 1.0f / fmaxf(cnt[n], 1.0f);
    Feat[(long)n * FEAT_COLS + outcol + 0] = acc[2 * (long)n + 0] * inv;
    Feat[(long)n * FEAT_COLS + outcol + 1] = acc[2 * (long)n + 1] * inv;
}

// Pack W1h (rows 128..269) and W1t (rows 398..539) into bf16 WMMA-B fragment
// order: Bpack[mat][kstep][coltile][lane][i], 16-bit B 32x16 layout:
// lane holds column (coltile*16 + lane&15); lanes 0-15 hold K=ks*32+{0..7,16..23},
// lanes 16-31 hold K=ks*32+{8..15,24..31}.
__global__ __launch_bounds__(256)
void packB_kernel(const float* __restrict__ W1, __bf16* __restrict__ Bpack) {
    int idx = blockIdx.x * 256 + threadIdx.x;
    const int TOTAL = 2 * KSTEPS * 8 * 32 * 16;
    if (idx >= TOTAL) return;
    int i    = idx & 15;
    int lane = (idx >> 4) & 31;
    int ct   = (idx >> 9) & 7;
    int rem  = idx >> 12;
    int ks   = rem % KSTEPS;
    int mat  = rem / KSTEPS;
    int K    = ks * 32 + ((i & 8) ? 16 : 0) + ((lane & 16) ? 8 : 0) + (i & 7);
    int col  = ct * 16 + (lane & 15);
    float v = 0.0f;
    if (K < KREAL) v = W1[(long)((mat ? 398 : 128) + K) * EMB + col];
    Bpack[idx] = (__bf16)v;
}

// PR2[r][j] = b1[j] + sum_k q[k]*W1[k][j] + sum_k rel[r][k]*W1[270+k][j]
__global__ __launch_bounds__(128)
void rel_kernel(const float* __restrict__ q, const float* __restrict__ rel,
                const float* __restrict__ W1, const float* __restrict__ b1,
                float* __restrict__ PR2) {
    int r = blockIdx.x;
    int j = threadIdx.x;
    float acc = b1[j];
    #pragma unroll 4
    for (int k = 0; k < EMB; ++k) acc += q[k] * W1[(long)k * EMB + j];
    const float* rv = rel + (long)r * EMB;
    #pragma unroll 4
    for (int k = 0; k < EMB; ++k) acc += rv[k] * W1[(long)(270 + k) * EMB + j];
    PR2[(long)r * EMB + j] = acc;
}

// ------------------------------------------------------------- node GEMMs
// One block = one 16-node tile; 4 waves each own 32 output columns of both
// PH and PT. A tile staged in LDS in 16-bit A 16x32 fragment order
// (lanes 0-15: M=row, K in {0..7,16..23}; lanes 16-31: K in {8..15,24..31}).
__global__ __launch_bounds__(128)
void gemm_kernel(const float* __restrict__ entity, const float* __restrict__ nontext,
                 const float* __restrict__ Feat, const __bf16* __restrict__ Bpack,
                 float* __restrict__ PH, float* __restrict__ PT, int N, int NTEXT) {
    __shared__ __bf16 sA[KSTEPS * 32 * 16];
    const int tile = blockIdx.x;
    const int tid  = threadIdx.x;

    // Stage A tile (16 rows x KPAD cols of h_e, converted to bf16) into LDS.
    for (int idx = tid; idx < 16 * KPAD; idx += 128) {
        int nl = idx / KPAD;
        int k  = idx - nl * KPAD;
        int n  = tile * 16 + nl;
        float v = 0.0f;
        if (n < N) {
            if (k < EMB)                    v = (n < NTEXT) ? entity[(long)n * EMB + k] : nontext[k];
            else if (k < EMB + FEAT_COLS)   v = Feat[(long)n * FEAT_COLS + (k - EMB)];
        }
        int ks = k >> 5, kk = k & 31;
        int lane_t = nl + ((kk & 8) ? 16 : 0);
        int ii     = ((kk & 16) ? 8 : 0) + (kk & 7);
        sA[(ks * 32 + lane_t) * 16 + ii] = (__bf16)v;
    }
    __syncthreads();

    const int wave = tid >> 5;
    const int lane = tid & 31;
    const int ct0 = wave * 2, ct1 = wave * 2 + 1;
    const v16bf* sAv = reinterpret_cast<const v16bf*>(sA);
    const v16bf* Bv  = reinterpret_cast<const v16bf*>(Bpack);

    v8f aH0 = {}, aH1 = {}, aT0 = {}, aT1 = {};
    #pragma unroll
    for (int ks = 0; ks < KSTEPS; ++ks) {
        v16bf a   = sAv[ks * 32 + lane];
        v16bf bh0 = Bv[(((0 * KSTEPS + ks) * 8 + ct0) * 32) + lane];
        v16bf bh1 = Bv[(((0 * KSTEPS + ks) * 8 + ct1) * 32) + lane];
        v16bf bt0 = Bv[(((1 * KSTEPS + ks) * 8 + ct0) * 32) + lane];
        v16bf bt1 = Bv[(((1 * KSTEPS + ks) * 8 + ct1) * 32) + lane];
        aH0 = __builtin_amdgcn_wmma_f32_16x16x32_bf16(false, a, false, bh0, (short)0, aH0, false, false);
        aH1 = __builtin_amdgcn_wmma_f32_16x16x32_bf16(false, a, false, bh1, (short)0, aH1, false, false);
        aT0 = __builtin_amdgcn_wmma_f32_16x16x32_bf16(false, a, false, bt0, (short)0, aT0, false, false);
        aT1 = __builtin_amdgcn_wmma_f32_16x16x32_bf16(false, a, false, bt1, (short)0, aT1, false, false);
    }

    // C/D layout: VGPR v, lanes 0-15 -> M=v, N=lane; lanes 16-31 -> M=v+8, N=lane-16.
    int mhi = (lane >= 16) ? 8 : 0;
    int nc  = lane & 15;
    #pragma unroll
    for (int v = 0; v < 8; ++v) {
        int n = tile * 16 + v + mhi;
        if (n < N) {
            PH[(long)n * EMB + ct0 * 16 + nc] = aH0[v];
            PH[(long)n * EMB + ct1 * 16 + nc] = aH1[v];
            PT[(long)n * EMB + ct0 * 16 + nc] = aT0[v];
            PT[(long)n * EMB + ct1 * 16 + nc] = aT1[v];
        }
    }
}

// --------------------------------------------------------------- edge pass
// out[e] = relu(PH[h] + PR2[r] + PT[t]) . W2 + b2 ; 4 lanes per edge,
// each lane covers 32 channels via float4 loads, shfl_xor reduction (wave32).
__global__ __launch_bounds__(256)
void edge_kernel(const int* __restrict__ h, const int* __restrict__ r, const int* __restrict__ t,
                 const float* __restrict__ PH, const float* __restrict__ PT,
                 const float* __restrict__ PR2, const float* __restrict__ W2,
                 const float* __restrict__ b2, float* __restrict__ out, int E) {
    int gid = blockIdx.x * 256 + threadIdx.x;
    int e   = gid >> 2;
    int sub = gid & 3;
    if (e >= E) return;
    int hi = h[e], ri = r[e], ti = t[e];
    const float4* ph = reinterpret_cast<const float4*>(PH  + (long)hi * EMB + sub * 32);
    const float4* pt = reinterpret_cast<const float4*>(PT  + (long)ti * EMB + sub * 32);
    const float4* pr = reinterpret_cast<const float4*>(PR2 + (long)ri * EMB + sub * 32);
    const float4* w2 = reinterpret_cast<const float4*>(W2  + sub * 32);
    float sum = 0.0f;
    #pragma unroll
    for (int j = 0; j < 8; ++j) {
        float4 a = ph[j], b = pr[j], c = pt[j], w = w2[j];
        float x;
        x = a.x + b.x + c.x; x = x > 0.0f ? x : 0.0f; sum += x * w.x;
        x = a.y + b.y + c.y; x = x > 0.0f ? x : 0.0f; sum += x * w.y;
        x = a.z + b.z + c.z; x = x > 0.0f ? x : 0.0f; sum += x * w.z;
        x = a.w + b.w + c.w; x = x > 0.0f ? x : 0.0f; sum += x * w.w;
    }
    sum += __shfl_xor(sum, 1, 32);
    sum += __shfl_xor(sum, 2, 32);
    if (sub == 0) out[e] = sum + b2[0];
}

// ---------------------------------------------------------------------------
extern "C" void kernel_launch(void* const* d_in, const int* in_sizes, int n_in,
                              void* d_out, int out_size, void* d_ws, size_t ws_size,
                              hipStream_t stream) {
    const int*   h_id    = (const int*)d_in[0];
    const int*   r_id    = (const int*)d_in[1];
    const int*   t_id    = (const int*)d_in[2];
    const float* q       = (const float*)d_in[3];
    const float* ent     = (const float*)d_in[4];
    const float* rel     = (const float*)d_in[6];
    const float* topic   = (const float*)d_in[7];
    const float* nontext = (const float*)d_in[8];
    const float* W1      = (const float*)d_in[9];
    const float* b1      = (const float*)d_in[10];
    const float* W2      = (const float*)d_in[11];
    const float* b2      = (const float*)d_in[12];

    const int E     = in_sizes[0];
    const int NTEXT = in_sizes[4] / EMB;
    const int NREL  = in_sizes[6] / EMB;
    const int N     = in_sizes[7] / 2;

    // workspace carve-up (256B aligned regions)
    char* w = (char*)d_ws;
    auto carve = [&](size_t bytes) { char* p = w; w += (bytes + 255) & ~(size_t)255; return p; };
    float*  PH    = (float*)carve((size_t)N * EMB * 4);       // 51.2 MB
    float*  PT    = (float*)carve((size_t)N * EMB * 4);       // 51.2 MB
    float*  Feat  = (float*)carve((size_t)N * FEAT_COLS * 4); //  6.4 MB
    float*  cntF  = (float*)carve((size_t)N * 4);
    float*  cntR  = (float*)carve((size_t)N * 4);
    float*  acc   = (float*)carve((size_t)N * 2 * 4);
    float*  PR2   = (float*)carve((size_t)NREL * EMB * 4);
    __bf16* Bpack = (__bf16*)carve((size_t)2 * KSTEPS * 8 * 32 * 16 * 2);

    const int nbN = (N + 255) / 256;
    const int nbE = (E + 255) / 256;

    hipMemsetAsync(cntF, 0, (size_t)N * 4, stream);
    hipMemsetAsync(cntR, 0, (size_t)N * 4, stream);
    init_feat_kernel<<<nbN, 256, 0, stream>>>(topic, Feat, N);
    count_kernel<<<nbE, 256, 0, stream>>>(h_id, t_id, cntF, cntR, E);

    // forward rounds: propagate h -> t  (input cols 0,2,4 -> output cols 2,4,6)
    const int inF[3] = {0, 2, 4};
    for (int rd = 0; rd < 3; ++rd) {
        hipMemsetAsync(acc, 0, (size_t)N * 2 * 4, stream);
        scatter_kernel<<<nbE, 256, 0, stream>>>(h_id, t_id, Feat, acc, inF[rd], E);
        finalize_kernel<<<nbN, 256, 0, stream>>>(acc, cntF, Feat, 2 + 2 * rd, N);
    }
    // reverse rounds: propagate t -> h  (input cols 0,8,10 -> output cols 8,10,12)
    const int inR[3] = {0, 8, 10};
    for (int rd = 0; rd < 3; ++rd) {
        hipMemsetAsync(acc, 0, (size_t)N * 2 * 4, stream);
        scatter_kernel<<<nbE, 256, 0, stream>>>(t_id, h_id, Feat, acc, inR[rd], E);
        finalize_kernel<<<nbN, 256, 0, stream>>>(acc, cntR, Feat, 8 + 2 * rd, N);
    }

    packB_kernel<<<(2 * KSTEPS * 8 * 32 * 16 + 255) / 256, 256, 0, stream>>>(W1, Bpack);
    rel_kernel<<<NREL, 128, 0, stream>>>(q, rel, W1, b1, PR2);

    const int tiles = (N + 15) / 16;
    gemm_kernel<<<tiles, 128, 0, stream>>>(ent, nontext, Feat, Bpack, PH, PT, N, NTEXT);

    const int ebl = ((E * 4) + 255) / 256;
    edge_kernel<<<ebl, 256, 0, stream>>>(h_id, r_id, t_id, PH, PT, PR2, W2, b2,
                                         (float*)d_out, E);
}
